// Connection_geognn_17076789969601
// MI455X (gfx1250) — matrix operand order repaired and checked by reading, not compiled, not run on hardware
//
#include <hip/hip_runtime.h>

typedef _Float16 f16;
typedef __attribute__((ext_vector_type(4)))  _Float16 v4h;
typedef __attribute__((ext_vector_type(8)))  _Float16 v8h;
typedef __attribute__((ext_vector_type(16))) _Float16 v16h;
typedef __attribute__((ext_vector_type(8)))  float    v8f;
typedef __attribute__((ext_vector_type(4)))  float    v4f;

// exact parameter types for the async-LDS builtin (per hipcc diagnostic):
typedef int v4i_g __attribute__((vector_size(16)));
typedef __attribute__((address_space(1))) v4i_g* as1_v4i;   // global
typedef __attribute__((address_space(3))) v4i_g* as3_v4i;   // LDS

#define DD      64      // D
#define TWO_D   128     // 2*D
#define OO2     256     // RANK*D
#define NROWS   262144
#define ROWS_PER_BLOCK 128          // 8 waves * 16 rows
#define NCHUNK  (NROWS / ROWS_PER_BLOCK)   // 2048
#define SMEM_BYTES ((128*64 + 64*128 + 256*128 + 128*256) * 2 + (128 + 256) * 4 + 8 * 16384)

#if __has_builtin(__builtin_amdgcn_global_load_async_to_lds_b128)
#define HAVE_ASYNC_LDS 1
#else
#define HAVE_ASYNC_LDS 0
#endif

__device__ __forceinline__ void wait_async0() {
#if __has_builtin(__builtin_amdgcn_s_wait_asynccnt)
    __builtin_amdgcn_s_wait_asynccnt(0);
#else
    asm volatile("s_wait_asynccnt 0" ::: "memory");
#endif
}

__device__ __forceinline__ float fast_tanh(float x) {
#if __has_builtin(__builtin_amdgcn_tanhf)
    return __builtin_amdgcn_tanhf(x);          // single v_tanh_f32 (CDNA5 TRANS op)
#elif __has_builtin(__builtin_amdgcn_rcpf)
    float e = __expf(2.0f * x);
    return 1.0f - 2.0f * __builtin_amdgcn_rcpf(e + 1.0f);
#else
    float e = __expf(2.0f * x);
    return 1.0f - 2.0f / (e + 1.0f);
#endif
}

// A-operand (16xK f16, row-within-lane): lane l -> M=l&15, K chunks {kb+8*(l>>4) .. +7} and {+16 ..}
__device__ __forceinline__ v16h ldA(const f16* base, int stride, int lane, int kb) {
    int m  = lane & 15;
    int k0 = kb + ((lane >> 4) << 3);
    v8h lo = *(const v8h*)(base + m * stride + k0);
    v8h hi = *(const v8h*)(base + m * stride + k0 + 16);
    return __builtin_shufflevector(lo, hi, 0,1,2,3,4,5,6,7,8,9,10,11,12,13,14,15);
}

// B-operand (Kx16 f16, col-within-lane): lane l -> N=n, K chunks {kb+8*(l>>4) .. +7} and {+16 ..}
__device__ __forceinline__ v16h ldB(const f16* base, int stride, int lane, int n, int kb) {
    int k0 = kb + ((lane >> 4) << 3);
    v8h lo = *(const v8h*)(base + n * stride + k0);
    v8h hi = *(const v8h*)(base + n * stride + k0 + 16);
    return __builtin_shufflevector(lo, hi, 0,1,2,3,4,5,6,7,8,9,10,11,12,13,14,15);
}

#define WMMA(a, b, c) __builtin_amdgcn_wmma_f32_16x16x32_f16(false, (a), false, (b), (short)0, (c), false, false)

__global__ __launch_bounds__(256) void geognn_grad_kernel(
    const float* __restrict__ input_,
    const float* __restrict__ W1, const float* __restrict__ b1,
    const float* __restrict__ W2, const float* __restrict__ b2,
    float* __restrict__ out)
{
    extern __shared__ char smem[];
    f16*   sW1  = (f16*)smem;             // [128][64]  forward B of GEMM1
    f16*   sW1T = sW1  + 128*64;          // [64][128]  backward B of GEMM4
    f16*   sW2  = sW1T + 64*128;          // [256][128] forward B of GEMM2
    f16*   sW2T = sW2  + 256*128;         // [128][256] backward B of GEMM3
    float* sB1  = (float*)(sW2T + 128*256);
    float* sB2  = sB1 + 128;
    char*  stg  = (char*)(sB2 + 256);

    const int tid  = threadIdx.x;
    const int wave = tid >> 5;
    const int lane = tid & 31;
    const int nlo   = lane & 15;
    const int nhalf = lane >> 4;

    // per-wave private staging (16 KB each)
    char*  st  = stg + wave * 16384;
    f16*   sM  = (f16*)st;             // [16][256] f16: M, then dz2 in-place
    float* sV  = (float*)(st + 8192);  // [16][64]  f32
    f16*   sXH = (f16*)(st + 12288);   // [16][64] X, then [16][128] H, then dz1

    // ---- stage weights (f32 -> f16, both orientations) ----
    for (int i = tid; i < 128*64; i += 256) {
        f16 h = (f16)W1[i];
        int r = i >> 6, c = i & 63;
        sW1[i] = h; sW1T[c * 128 + r] = h;
    }
    for (int i = tid; i < 256*128; i += 256) {
        f16 h = (f16)W2[i];
        int r = i >> 7, c = i & 127;
        sW2[i] = h; sW2T[c * 256 + r] = h;
    }
    if (tid < 128) sB1[tid] = b1[tid];
    sB2[tid] = b2[tid];
    __syncthreads();

    for (int chunk = blockIdx.x; chunk < NCHUNK; chunk += (int)gridDim.x) {
        const int row0 = chunk * ROWS_PER_BLOCK + wave * 16;

        // ---- prefetch next chunk's rows (branchless; clamps to current chunk) ----
        {
            int nextc = chunk + (int)gridDim.x;
            if (nextc >= NCHUNK) nextc = chunk;           // cndmask, no branch
            const char* pf = (const char*)(input_ +
                (size_t)(nextc * ROWS_PER_BLOCK + wave * 16) * TWO_D);
            __builtin_prefetch(pf + lane * 256, 0, 1);
            __builtin_prefetch(pf + lane * 256 + 128, 0, 1);
        }

        // ---- Phase 1a: V (16x64 f32) -> LDS via async DMA (no VGPR transit) ----
#if HAVE_ASYNC_LDS
        #pragma unroll 2
        for (int j = 0; j < 8; ++j) {
            int flat = j * 32 + lane;          // 0..255 float4 tiles
            int r = flat >> 4;
            int c = (flat & 15) * 4;
            float* gsrc = const_cast<float*>(input_) + (size_t)(row0 + r) * TWO_D + DD + c;
            __builtin_amdgcn_global_load_async_to_lds_b128(
                (as1_v4i)gsrc, (as3_v4i)(sV + r * DD + c), 0, 0);
        }
#else
        #pragma unroll 2
        for (int j = 0; j < 8; ++j) {
            int flat = j * 32 + lane;
            int r = flat >> 4;
            int c = (flat & 15) * 4;
            v4f d = *(const v4f*)(input_ + (size_t)(row0 + r) * TWO_D + DD + c);
            *(v4f*)(sV + r * DD + c) = d;
        }
#endif

        // ---- Phase 1b: X (16x64 f32) -> f16 LDS, coalesced ----
        #pragma unroll 2
        for (int j = 0; j < 8; ++j) {
            int flat = j * 32 + lane;
            int r = flat >> 4;
            int c = (flat & 15) * 4;
            v4f d = *(const v4f*)(input_ + (size_t)(row0 + r) * TWO_D + c);
            v4h hh; hh[0] = (f16)d[0]; hh[1] = (f16)d[1]; hh[2] = (f16)d[2]; hh[3] = (f16)d[3];
            *(v4h*)(sXH + r * DD + c) = hh;
        }

        // ---- Phase 2: Z1 = X * W1^T (16x64 @ 64x128); H = tanh(Z1 + b1) ----
        // groups of 4 tiles: pass A = all WMMAs (deep ds->wmma pipelining),
        // pass B = activations (also spaces WMMA->VALU past hazard window)
        v16h aX0 = ldA(sXH, DD, lane, 0);
        v16h aX1 = ldA(sXH, DD, lane, 32);
        float Hc[8][8];   // H in C-layout registers (matches GEMM3 output layout)
        #pragma unroll
        for (int tg = 0; tg < 8; tg += 4) {
            v8f cg[4];
            #pragma unroll
            for (int q = 0; q < 4; ++q) {
                int t = tg + q;
                v8f c = {};
                c = WMMA(aX0, ldB(sW1, DD, lane, t*16 + nlo, 0),  c);
                c = WMMA(aX1, ldB(sW1, DD, lane, t*16 + nlo, 32), c);
                cg[q] = c;
            }
            #pragma unroll
            for (int q = 0; q < 4; ++q) {
                int t = tg + q;
                float bias = sB1[t*16 + nlo];
                float zv[8], hv[8];
                #pragma unroll
                for (int r = 0; r < 8; ++r) zv[r] = cg[q][r] + bias;
                #pragma unroll
                for (int r = 0; r < 8; ++r) hv[r] = fast_tanh(zv[r]);
                f16* hdst = sXH + (8*nhalf) * TWO_D + t*16 + nlo;
                #pragma unroll
                for (int r = 0; r < 8; ++r) Hc[t][r] = hv[r];
                #pragma unroll
                for (int r = 0; r < 8; ++r) hdst[r * TWO_D] = (f16)hv[r];
            }
        }

        // ---- Phase 3: Z2 = H * W2^T (16x128 @ 128x256); M = tanh(Z2 + b2) -> LDS ----
        v16h aH[4];
        #pragma unroll
        for (int ks = 0; ks < 4; ++ks) aH[ks] = ldA(sXH, TWO_D, lane, ks*32);
        #pragma unroll
        for (int tg = 0; tg < 16; tg += 4) {
            v8f cg[4];
            #pragma unroll
            for (int q = 0; q < 4; ++q) {
                int t = tg + q;
                v8f c = {};
                #pragma unroll
                for (int ks = 0; ks < 4; ++ks)
                    c = WMMA(aH[ks], ldB(sW2, TWO_D, lane, t*16 + nlo, ks*32), c);
                cg[q] = c;
            }
            #pragma unroll
            for (int q = 0; q < 4; ++q) {
                int t = tg + q;
                float bias = sB2[t*16 + nlo];
                float zv[8], hv[8];
                #pragma unroll
                for (int r = 0; r < 8; ++r) zv[r] = cg[q][r] + bias;
                #pragma unroll
                for (int r = 0; r < 8; ++r) hv[r] = fast_tanh(zv[r]);
                f16* mdst = sM + (8*nhalf) * OO2 + t*16 + nlo;
                #pragma unroll
                for (int r = 0; r < 8; ++r) mdst[r * OO2] = (f16)hv[r];
            }
        }

        // ---- Phase 4: per-row middle: y, dv (out), dz2 (in-place over M) ----
        {
#if HAVE_ASYNC_LDS
            wait_async0();                     // V DMA must have landed in LDS
#endif
            const int   row   = nlo;           // 2 lanes per row (nhalf = d-range half)
            const int   dbase = nhalf * 32;
            const f16*  mrow  = sM + row * OO2;
            const float* vrow = sV + row * DD;

            float y[4] = {0.f, 0.f, 0.f, 0.f};
            #pragma unroll
            for (int d4 = 0; d4 < 32; d4 += 4) {
                v4f v4 = *(const v4f*)(vrow + dbase + d4);
                v8h ma = *(const v8h*)(mrow + 4*(dbase + d4));
                v8h mb = *(const v8h*)(mrow + 4*(dbase + d4 + 2));
                y[0] += v4[0]*(float)ma[0] + v4[1]*(float)ma[4] + v4[2]*(float)mb[0] + v4[3]*(float)mb[4];
                y[1] += v4[0]*(float)ma[1] + v4[1]*(float)ma[5] + v4[2]*(float)mb[1] + v4[3]*(float)mb[5];
                y[2] += v4[0]*(float)ma[2] + v4[1]*(float)ma[6] + v4[2]*(float)mb[2] + v4[3]*(float)mb[6];
                y[3] += v4[0]*(float)ma[3] + v4[1]*(float)ma[7] + v4[2]*(float)mb[3] + v4[3]*(float)mb[7];
            }
            float yc[4];
            #pragma unroll
            for (int rr = 0; rr < 4; ++rr) {
                y[rr] += __shfl_xor(y[rr], 16, 32);       // combine the two d-halves
                yc[rr] = 2.0f * y[rr];
            }

            float* orow = out + (size_t)(row0 + row) * TWO_D;
            f16*   zrow = sM + row * OO2;
            #pragma unroll
            for (int d4 = 0; d4 < 32; d4 += 4) {
                v4f v4 = *(const v4f*)(vrow + dbase + d4);
                v8h ma = *(const v8h*)(mrow + 4*(dbase + d4));
                v8h mb = *(const v8h*)(mrow + 4*(dbase + d4 + 2));
                v8h za, zb;
                v4f dv4;
                float acc, mv;
                acc = 0.f;
                #pragma unroll
                for (int rr = 0; rr < 4; ++rr) {
                    mv = (float)ma[rr];
                    za[rr] = (f16)(yc[rr] * v4[0] * (1.0f - mv*mv));
                    acc += yc[rr] * mv;
                }
                dv4[0] = -acc;
                acc = 0.f;
                #pragma unroll
                for (int rr = 0; rr < 4; ++rr) {
                    mv = (float)ma[4 + rr];
                    za[4 + rr] = (f16)(yc[rr] * v4[1] * (1.0f - mv*mv));
                    acc += yc[rr] * mv;
                }
                dv4[1] = -acc;
                acc = 0.f;
                #pragma unroll
                for (int rr = 0; rr < 4; ++rr) {
                    mv = (float)mb[rr];
                    zb[rr] = (f16)(yc[rr] * v4[2] * (1.0f - mv*mv));
                    acc += yc[rr] * mv;
                }
                dv4[2] = -acc;
                acc = 0.f;
                #pragma unroll
                for (int rr = 0; rr < 4; ++rr) {
                    mv = (float)mb[4 + rr];
                    zb[4 + rr] = (f16)(yc[rr] * v4[3] * (1.0f - mv*mv));
                    acc += yc[rr] * mv;
                }
                dv4[3] = -acc;
                *(v8h*)(zrow + 4*(dbase + d4))     = za;   // dz2, b128 store
                *(v8h*)(zrow + 4*(dbase + d4 + 2)) = zb;   // dz2, b128 store
                *(v4f*)(orow + DD + dbase + d4)    = dv4;  // dv output
            }
        }

        // ---- Phase 5: dh = dz2 * W2 (16x256 @ 256x128); dz1 = dh * (1 - H^2) ----
        v16h aZ[8];
        #pragma unroll
        for (int ks = 0; ks < 8; ++ks) aZ[ks] = ldA(sM, OO2, lane, ks*32);
        #pragma unroll
        for (int tg = 0; tg < 8; tg += 4) {
            v8f cg[4];
            #pragma unroll
            for (int q = 0; q < 4; ++q) {
                int t = tg + q;
                v8f c = {};
                #pragma unroll
                for (int ks = 0; ks < 8; ++ks)
                    c = WMMA(aZ[ks], ldB(sW2T, OO2, lane, t*16 + nlo, ks*32), c);
                cg[q] = c;
            }
            #pragma unroll
            for (int q = 0; q < 4; ++q) {
                int t = tg + q;
                float dz[8];
                #pragma unroll
                for (int r = 0; r < 8; ++r) {
                    float h = Hc[t][r];
                    dz[r] = cg[q][r] * (1.0f - h*h);
                }
                f16* ddst = sXH + (8*nhalf) * TWO_D + t*16 + nlo;
                #pragma unroll
                for (int r = 0; r < 8; ++r) ddst[r * TWO_D] = (f16)dz[r];
            }
        }

        // ---- Phase 6: dx = dz1 * W1 (16x128 @ 128x64) -> global ----
        v16h aD[4];
        #pragma unroll
        for (int ks = 0; ks < 4; ++ks) aD[ks] = ldA(sXH, TWO_D, lane, ks*32);
        #pragma unroll
        for (int t = 0; t < 4; ++t) {
            v8f c = {};
            #pragma unroll
            for (int ks = 0; ks < 4; ++ks)
                c = WMMA(aD[ks], ldB(sW1T, TWO_D, lane, t*16 + nlo, ks*32), c);
            #pragma unroll
            for (int r = 0; r < 8; ++r)
                out[(size_t)(row0 + r + 8*nhalf) * TWO_D + t*16 + nlo] = c[r];
        }
    }
}

extern "C" void kernel_launch(void* const* d_in, const int* in_sizes, int n_in,
                              void* d_out, int out_size, void* d_ws, size_t ws_size,
                              hipStream_t stream) {
    (void)in_sizes; (void)n_in; (void)out_size; (void)d_ws; (void)ws_size;
    // d_in order: t(unused), input_, W1, b1, W2, b2 — all float32
    const float* input_ = (const float*)d_in[1];
    const float* W1     = (const float*)d_in[2];
    const float* b1     = (const float*)d_in[3];
    const float* W2     = (const float*)d_in[4];
    const float* b2     = (const float*)d_in[5];
    float* out = (float*)d_out;

    geognn_grad_kernel<<<dim3(1024), dim3(256), SMEM_BYTES, stream>>>(
        input_, W1, b1, W2, b2, out);
}
